// force_net_10694468567092
// MI455X (gfx1250) — matrix operand order repaired and checked by reading, not compiled, not run on hardware
//
#include <hip/hip_runtime.h>
#include <math.h>

// ---------------------------------------------------------------------------
// MI455X (gfx1250) transformer forward.
// All GEMMs, attention, and the score head via v_wmma_f32_16x16x32_bf16.
// Flash attention (no MxM materialization); score head via WMMA + elementwise
// LayerNorm across head-indexed accumulator registers.
// ---------------------------------------------------------------------------

typedef __attribute__((ext_vector_type(16))) __bf16 v16bf;
typedef __attribute__((ext_vector_type(8)))  float  v8f;
typedef __attribute__((ext_vector_type(4)))  float  v4f;

#define DIM    256
#define DFFDIM 1024
#define HN     8
#define DKD    32
#define MSEQ   1024
#define BNUM   4
#define RTOT   (BNUM * MSEQ)
#define LLAY   4
#define INVSQRT_DK 0.17677669529663687f  // 1/sqrt(32) == sqrt(2/NFREQ)

static __device__ __forceinline__ __bf16 f2bf(float f) {
  unsigned u = __builtin_bit_cast(unsigned, f);
  unsigned r = u + 0x7FFFu + ((u >> 16) & 1u);        // round-to-nearest-even
  unsigned short h = (unsigned short)(r >> 16);
  return __builtin_bit_cast(__bf16, h);
}
static __device__ __forceinline__ float bf2f(__bf16 b) {
  unsigned short h = __builtin_bit_cast(unsigned short, b);
  unsigned u = ((unsigned)h) << 16;
  return __builtin_bit_cast(float, u);
}

union ABits { v16bf v; v4f f[2]; };

// A fragment: 16x32 bf16 tile from a row-major matrix, row stride lda (elems).
// Layout: lanes 0-15 row=lane, elems0-15 = K 0..15;
// lanes 16-31 same rows, K 16..31.  => two contiguous 16B loads.
static __device__ __forceinline__ v16bf load_a_frag(const __bf16* base, int lda, int lane) {
  const int row = lane & 15, hf = lane >> 4;
  const __bf16* p = base + (size_t)row * lda + 16 * hf;
  ABits u;
  u.f[0] = *(const v4f*)p;
  u.f[1] = *(const v4f*)(p + 8);
  return u.v;
}
// B fragment: 32x16 bf16, source stored N-major (each output-col row holds K
// contiguous), stride ldb. Lanes 0-15 col=lane K 0..15; lanes 16-31 K 16..31.
// => one contiguous 32B load per lane.
static __device__ __forceinline__ v16bf load_b_frag(const __bf16* base, int ldb, int lane) {
  const int n = lane & 15, hf = lane >> 4;
  return *(const v16bf*)(base + (size_t)n * ldb + 16 * hf);
}

static __device__ __forceinline__ v8f wmma_bf16(v16bf a, v16bf b, v8f c) {
  return __builtin_amdgcn_wmma_f32_16x16x32_bf16(false, a, false, b, (short)0, c, false, false);
}

// ---------------------------------------------------------------------------
// Weight convert+transpose: src fp32 (count, K, N) -> dst bf16 (count, N, K)
// ---------------------------------------------------------------------------
__global__ void k_wt(const float* __restrict__ W, __bf16* __restrict__ Wt,
                     int K, int N, int count) {
  size_t idx = (size_t)blockIdx.x * blockDim.x + threadIdx.x;
  size_t tot = (size_t)count * K * N;
  if (idx >= tot) return;
  size_t kn  = (size_t)K * N;
  int s   = (int)(idx / kn);
  size_t rem = idx - (size_t)s * kn;
  int n = (int)(rem % N), k = (int)(rem / N);
  Wt[(size_t)s * kn + (size_t)n * K + k] = f2bf(W[idx]);
}

// ---------------------------------------------------------------------------
// Embedding: x = X[:,:,6:] @ char_W + char_b + fourier_pe
// ---------------------------------------------------------------------------
__global__ void k_embed(const float* __restrict__ X, const float* __restrict__ fW,
                        const float* __restrict__ fb, const float* __restrict__ cW,
                        const float* __restrict__ cb, float* __restrict__ x) {
  int idx = blockIdx.x * blockDim.x + threadIdx.x;
  if (idx >= RTOT * DIM) return;
  int d = idx & (DIM - 1);
  int r = idx >> 8;
  const float* xr = X + (size_t)r * 70;
  float p0 = xr[0], p1 = xr[1], p2 = xr[2];
  int f = d & 127;
  float proj = p0 * fW[f * 3 + 0] + p1 * fW[f * 3 + 1] + p2 * fW[f * 3 + 2] + fb[f];
  float pe = ((d < 128) ? cosf(proj) : sinf(proj)) * INVSQRT_DK; // sqrt(2/64)
  float acc = cb[d];
  #pragma unroll 8
  for (int j = 0; j < 64; ++j) acc += xr[6 + j] * cW[j * DIM + d];
  x[idx] = pe + acc;
}

// ---------------------------------------------------------------------------
// LayerNorm over D=256: one wave per row, emits bf16 rows for WMMA A-frags.
// ---------------------------------------------------------------------------
__global__ __launch_bounds__(256)
void k_ln(const float* __restrict__ X, const float* __restrict__ g,
          const float* __restrict__ bta, __bf16* __restrict__ Y) {
  int row = blockIdx.x * 8 + (threadIdx.x >> 5);
  if (row >= RTOT) return;
  int lane = threadIdx.x & 31;
  const v4f* xp = (const v4f*)(X + (size_t)row * DIM);
  v4f a = xp[lane * 2], b = xp[lane * 2 + 1];
  float s = a.x + a.y + a.z + a.w + b.x + b.y + b.z + b.w;
  #pragma unroll
  for (int m = 1; m < 32; m <<= 1) s += __shfl_xor(s, m, 32);
  float mu = s * (1.0f / DIM);
  float vals[8] = {a.x, a.y, a.z, a.w, b.x, b.y, b.z, b.w};
  float vs = 0.f;
  #pragma unroll
  for (int j = 0; j < 8; ++j) { float t = vals[j] - mu; vs += t * t; }
  #pragma unroll
  for (int m = 1; m < 32; m <<= 1) vs += __shfl_xor(vs, m, 32);
  float inv = rsqrtf(vs * (1.0f / DIM) + 1e-5f);
  int col = lane * 8;
  union { __bf16 h[8]; v4f f; } o;
  #pragma unroll
  for (int j = 0; j < 8; ++j)
    o.h[j] = f2bf((vals[j] - mu) * inv * g[col + j] + bta[col + j]);
  *(v4f*)(Y + (size_t)row * DIM + col) = o.f;
}

// ---------------------------------------------------------------------------
// Generic WMMA GEMM: C(R,N) = A(R,K)bf16 @ Wt(N,K)bf16 + bias [+GELU][+resid]
// 16x64 tile per wave (A-fragment reused across 4 B-fragments / 4 WMMAs per
// K-step); 4 waves per block cover 256 output columns.
// VT: scatter-store V transposed as (b,h,dk,m) bf16 for the flash PV stage.
// ---------------------------------------------------------------------------
template <bool GELU, bool RESID, bool OUTB, bool VT>
__global__ __launch_bounds__(128)
void k_gemm(const __bf16* __restrict__ A, const __bf16* __restrict__ Bt,
            const float* __restrict__ bias, const float* __restrict__ Res,
            float* __restrict__ Cf, __bf16* __restrict__ Cb,
            int R, int K, int N) {
  const int lane = threadIdx.x & 31;
  const int wv   = threadIdx.x >> 5;
  const int hf   = lane >> 4, l15 = lane & 15;
  const int ct = (blockIdx.x * 4 + wv) * 64;
  const int rt = blockIdx.y * 16;
  if (ct >= N || rt >= R) return;
  const __bf16* arow = A + (size_t)(rt + l15) * K;
  const __bf16* brow0 = Bt + (size_t)(ct + l15) * K + 16 * hf;
  v8f c0 = {0.f,0.f,0.f,0.f,0.f,0.f,0.f,0.f};
  v8f c1 = c0, c2 = c0, c3 = c0;
  const size_t bstep = (size_t)16 * K;   // 16 output cols ahead in N-major Bt
  for (int k0 = 0; k0 < K; k0 += 32) {
    if (k0 + 64 <= K) {                  // stream-ahead -> global_prefetch_b8
      __builtin_prefetch(arow + k0 + 32, 0, 1);
      __builtin_prefetch(brow0 + k0 + 32, 0, 1);
    }
    ABits ua;
    const __bf16* ap = arow + k0 + 16 * hf;
    ua.f[0] = *(const v4f*)ap;
    ua.f[1] = *(const v4f*)(ap + 8);
    v16bf b0 = *(const v16bf*)(brow0 + k0);
    v16bf b1 = *(const v16bf*)(brow0 + bstep + k0);
    v16bf b2 = *(const v16bf*)(brow0 + 2 * bstep + k0);
    v16bf b3 = *(const v16bf*)(brow0 + 3 * bstep + k0);
    c0 = wmma_bf16(ua.v, b0, c0);
    c1 = wmma_bf16(ua.v, b1, c1);
    c2 = wmma_bf16(ua.v, b2, c2);
    c3 = wmma_bf16(ua.v, b3, c3);
  }
  v8f accs[4] = {c0, c1, c2, c3};
  #pragma unroll
  for (int t = 0; t < 4; ++t) {
    if (ct + 16 * t >= N) break;
    #pragma unroll
    for (int r = 0; r < 8; ++r) {
      int row = rt + r + 8 * hf;
      int col = ct + 16 * t + l15;
      float v = accs[t][r] + bias[col];
      if (GELU) v = 0.5f * v * (1.0f + erff(v * 0.7071067811865476f));
      if (RESID) v += Res[(size_t)row * N + col];
      if (VT) { // (b,h,dk,m) layout, key index contiguous
        int b_ = row >> 10, m_ = row & (MSEQ - 1);
        int h_ = col >> 5,  dk = col & 31;
        Cb[((size_t)((b_ * HN + h_) * DKD + dk)) * MSEQ + m_] = f2bf(v);
      } else if (OUTB) {
        Cb[(size_t)row * N + col] = f2bf(v);
      } else {
        Cf[(size_t)row * N + col] = v;
      }
    }
  }
}

// ---------------------------------------------------------------------------
// Flash attention: one wave per (b, h, 16-row q tile). K-tiles of 32 keys.
// Scores via 2x WMMA, online softmax (16-lane shfl reductions), P through
// LDS to re-fragment as A operand, PV via 2x WMMA against transposed V.
// ---------------------------------------------------------------------------
__global__ __launch_bounds__(32)
void k_attn(const __bf16* __restrict__ Qb, const __bf16* __restrict__ Kb,
            const __bf16* __restrict__ Vt, const int* __restrict__ mask,
            __bf16* __restrict__ Att) {
  const int bh = blockIdx.x, b = bh >> 3, h = bh & 7;
  const int qt = blockIdx.y * 16;
  const int lane = threadIdx.x & 31;
  const int hf = lane >> 4, l15 = lane & 15;
  __shared__ __bf16 P[16 * 32];

  const __bf16* qbase = Qb + ((size_t)(b * MSEQ + qt)) * DIM + h * DKD;
  v16bf aq = load_a_frag(qbase, DIM, lane);

  v8f acc0 = {0.f,0.f,0.f,0.f,0.f,0.f,0.f,0.f};
  v8f acc1 = {0.f,0.f,0.f,0.f,0.f,0.f,0.f,0.f};
  float mrow[8], lrow[8];
  #pragma unroll
  for (int r = 0; r < 8; ++r) { mrow[r] = -3.0e38f; lrow[r] = 0.f; }

  for (int kt = 0; kt < MSEQ; kt += 32) {
    v16bf bk0 = load_b_frag(Kb + ((size_t)(b * MSEQ + kt))      * DIM + h * DKD, DIM, lane);
    v16bf bk1 = load_b_frag(Kb + ((size_t)(b * MSEQ + kt + 16)) * DIM + h * DKD, DIM, lane);
    v8f s0 = {0.f,0.f,0.f,0.f,0.f,0.f,0.f,0.f};
    v8f s1 = {0.f,0.f,0.f,0.f,0.f,0.f,0.f,0.f};
    s0 = wmma_bf16(aq, bk0, s0);
    s1 = wmma_bf16(aq, bk1, s1);
    const int col0 = kt + l15, col1 = kt + 16 + l15;
    #pragma unroll
    for (int r = 0; r < 8; ++r) {
      int row = qt + r + 8 * hf;
      const int* mrowp = mask + ((size_t)b * MSEQ + row) * MSEQ;
      float v0 = s0[r] * INVSQRT_DK, v1 = s1[r] * INVSQRT_DK;
      if (mrowp[col0] == 0) v0 = -3.0e38f;
      if (mrowp[col1] == 0) v1 = -3.0e38f;
      float tm = fmaxf(v0, v1);
      #pragma unroll
      for (int m = 1; m < 16; m <<= 1) tm = fmaxf(tm, __shfl_xor(tm, m, 32));
      float nm = fmaxf(mrow[r], tm);
      float corr = __expf(mrow[r] - nm);
      float p0 = __expf(v0 - nm), p1 = __expf(v1 - nm);
      float ts = p0 + p1;
      #pragma unroll
      for (int m = 1; m < 16; m <<= 1) ts += __shfl_xor(ts, m, 32);
      lrow[r] = lrow[r] * corr + ts;
      mrow[r] = nm;
      acc0[r] *= corr; acc1[r] *= corr;
      P[(r + 8 * hf) * 32 + l15]      = f2bf(p0);
      P[(r + 8 * hf) * 32 + 16 + l15] = f2bf(p1);
    }
    v16bf ap = load_a_frag(P, 32, lane);  // ds_load re-fragmentation
    const __bf16* vb = Vt + ((size_t)(b * HN + h) * DKD) * MSEQ + kt;
    v16bf bv0 = load_b_frag(vb,             MSEQ, lane);  // dk 0..15
    v16bf bv1 = load_b_frag(vb + 16 * MSEQ, MSEQ, lane);  // dk 16..31
    acc0 = wmma_bf16(ap, bv0, acc0);
    acc1 = wmma_bf16(ap, bv1, acc1);
  }
  #pragma unroll
  for (int r = 0; r < 8; ++r) {
    int row = qt + r + 8 * hf;
    float inv = 1.0f / lrow[r];
    __bf16* op = Att + ((size_t)(b * MSEQ) + row) * DIM + h * DKD;
    op[l15]      = f2bf(acc0[r] * inv);
    op[16 + l15] = f2bf(acc1[r] * inv);
  }
}

// ---------------------------------------------------------------------------
// Score head, stage 1: t(q,k) = headproj(LN_H(q . k / sqrt(dk))).
// One wave per 16x16 (q,k) tile: 8 WMMAs (one per head).  The 8 head scores
// for a given (q,k) element live in the SAME lane across 8 accumulator
// fragments, so LN-over-heads + projection is pure elementwise VALU.
// ---------------------------------------------------------------------------
__global__ __launch_bounds__(32)
void k_scores(const __bf16* __restrict__ Qb, const __bf16* __restrict__ Kb,
              const float* __restrict__ snw, const float* __restrict__ snb,
              const float* __restrict__ sfw, const float* __restrict__ sfb,
              float* __restrict__ S) {
  const int bq = blockIdx.x;
  const int b  = bq >> 6;                 // M/16 == 64 tiles
  const int qt = (bq & 63) * 16;
  const int kt = blockIdx.y * 16;
  const int lane = threadIdx.x & 31;
  const int hf = lane >> 4, l15 = lane & 15;

  v8f sc[HN];
  #pragma unroll
  for (int h = 0; h < HN; ++h) {
    v16bf aq = load_a_frag(Qb + ((size_t)(b * MSEQ + qt)) * DIM + h * DKD, DIM, lane);
    v16bf bk = load_b_frag(Kb + ((size_t)(b * MSEQ + kt)) * DIM + h * DKD, DIM, lane);
    v8f z = {0.f,0.f,0.f,0.f,0.f,0.f,0.f,0.f};
    sc[h] = wmma_bf16(aq, bk, z);
  }
  float w0 = snw[0+0]; // force scalar loads once
  (void)w0;
  #pragma unroll
  for (int r = 0; r < 8; ++r) {
    float v[HN]; float mu = 0.f;
    #pragma unroll
    for (int h = 0; h < HN; ++h) { v[h] = sc[h][r] * INVSQRT_DK; mu += v[h]; }
    mu *= 0.125f;
    float var = 0.f;
    #pragma unroll
    for (int h = 0; h < HN; ++h) { float d = v[h] - mu; var += d * d; }
    float inv = rsqrtf(var * 0.125f + 1e-5f);
    float o = sfb[0];
    #pragma unroll
    for (int h = 0; h < HN; ++h)
      o += ((v[h] - mu) * inv * snw[h] + snb[h]) * sfw[h];
    int row = qt + r + 8 * hf;
    S[((size_t)b * MSEQ + row) * MSEQ + kt + l15] = o;
  }
}

// ---------------------------------------------------------------------------
// Score head, stage 2: symmetrize in place.  Exactly one writer per
// (q,k)/(k,q) pair -> race-free and deterministic.
// ---------------------------------------------------------------------------
__global__ void k_sym(float* __restrict__ S) {
  size_t idx = (size_t)blockIdx.x * blockDim.x + threadIdx.x;
  size_t tot = (size_t)BNUM * MSEQ * MSEQ;
  if (idx >= tot) return;
  int ki = (int)(idx & (MSEQ - 1));
  size_t t = idx >> 10;
  int qi = (int)(t & (MSEQ - 1));
  int b  = (int)(t >> 10);
  if (qi > ki) return;
  size_t i0 = ((size_t)b * MSEQ + qi) * MSEQ + ki;
  size_t i1 = ((size_t)b * MSEQ + ki) * MSEQ + qi;
  float a = 0.5f * (S[i0] + S[i1]);
  S[i0] = a;
  S[i1] = a;
}

// ---------------------------------------------------------------------------
extern "C" void kernel_launch(void* const* d_in, const int* in_sizes, int n_in,
                              void* d_out, int out_size, void* d_ws, size_t ws_size,
                              hipStream_t stream) {
  const float* X    = (const float*)d_in[0];
  const int*   mask = (const int*)  d_in[1];
  const float* fW   = (const float*)d_in[2];
  const float* fb   = (const float*)d_in[3];
  const float* cW   = (const float*)d_in[4];
  const float* cb   = (const float*)d_in[5];
  const float* Wq   = (const float*)d_in[6];
  const float* bq   = (const float*)d_in[7];
  const float* Wk   = (const float*)d_in[8];
  const float* bk   = (const float*)d_in[9];
  const float* Wv   = (const float*)d_in[10];
  const float* bv   = (const float*)d_in[11];
  const float* Wo   = (const float*)d_in[12];
  const float* bo   = (const float*)d_in[13];
  const float* ln1w = (const float*)d_in[14];
  const float* ln1b = (const float*)d_in[15];
  const float* ln2w = (const float*)d_in[16];
  const float* ln2b = (const float*)d_in[17];
  const float* lnfw = (const float*)d_in[18];
  const float* lnfb = (const float*)d_in[19];
  const float* Wup  = (const float*)d_in[20];
  const float* bup  = (const float*)d_in[21];
  const float* Wdn  = (const float*)d_in[22];
  const float* bdn  = (const float*)d_in[23];
  const float* Wfin = (const float*)d_in[24];
  const float* bfin = (const float*)d_in[25];
  const float* snw  = (const float*)d_in[26];
  const float* snb  = (const float*)d_in[27];
  const float* sfw  = (const float*)d_in[28];
  const float* sfb  = (const float*)d_in[29];
  (void)in_sizes; (void)n_in; (void)out_size; (void)ws_size;

  // workspace carve-up (256B aligned); total ~28.2 MB
  char* base = (char*)d_ws;
  size_t off = 0;
  auto carve = [&](size_t bytes) -> char* {
    off = (off + 255) & ~(size_t)255;
    char* p = base + off; off += bytes; return p;
  };
  const size_t DD = (size_t)DIM * DIM;
  __bf16* wtq = (__bf16*)carve(LLAY * DD * 2);
  __bf16* wtk = (__bf16*)carve(LLAY * DD * 2);
  __bf16* wtv = (__bf16*)carve(LLAY * DD * 2);
  __bf16* wto = (__bf16*)carve(LLAY * DD * 2);
  __bf16* wtu = (__bf16*)carve((size_t)LLAY * DIM * DFFDIM * 2);
  __bf16* wtd = (__bf16*)carve((size_t)LLAY * DFFDIM * DIM * 2);
  __bf16* wtf = (__bf16*)carve((size_t)DIM * 64 * 2);
  float*  xbuf = (float*) carve((size_t)RTOT * DIM * 4);
  __bf16* xn   = (__bf16*)carve((size_t)RTOT * DIM * 2);
  __bf16* qb   = (__bf16*)carve((size_t)RTOT * DIM * 2);
  __bf16* kb   = (__bf16*)carve((size_t)RTOT * DIM * 2);
  __bf16* vt   = (__bf16*)carve((size_t)RTOT * DIM * 2);
  __bf16* att  = (__bf16*)carve((size_t)RTOT * DIM * 2);
  __bf16* hdn  = (__bf16*)carve((size_t)RTOT * DFFDIM * 2);

  float* out_x = (float*)d_out;                       // (B, M, 64)
  float* out_s = (float*)d_out + (size_t)RTOT * 64;   // (B, M, M)

  const int TPB = 256;
  // weight convert+transpose to bf16 N-major
  {
    size_t n;
    n = (size_t)LLAY * DD;
    k_wt<<<dim3((unsigned)((n + TPB - 1) / TPB)), TPB, 0, stream>>>(Wq, wtq, DIM, DIM, LLAY);
    k_wt<<<dim3((unsigned)((n + TPB - 1) / TPB)), TPB, 0, stream>>>(Wk, wtk, DIM, DIM, LLAY);
    k_wt<<<dim3((unsigned)((n + TPB - 1) / TPB)), TPB, 0, stream>>>(Wv, wtv, DIM, DIM, LLAY);
    k_wt<<<dim3((unsigned)((n + TPB - 1) / TPB)), TPB, 0, stream>>>(Wo, wto, DIM, DIM, LLAY);
    n = (size_t)LLAY * DIM * DFFDIM;
    k_wt<<<dim3((unsigned)((n + TPB - 1) / TPB)), TPB, 0, stream>>>(Wup, wtu, DIM, DFFDIM, LLAY);
    k_wt<<<dim3((unsigned)((n + TPB - 1) / TPB)), TPB, 0, stream>>>(Wdn, wtd, DFFDIM, DIM, LLAY);
    n = (size_t)DIM * 64;
    k_wt<<<dim3((unsigned)((n + TPB - 1) / TPB)), TPB, 0, stream>>>(Wfin, wtf, DIM, 64, 1);
  }
  // embedding
  k_embed<<<dim3(RTOT * DIM / TPB), TPB, 0, stream>>>(X, fW, fb, cW, cb, xbuf);

  const dim3 lnGrid(RTOT / 8), lnBlk(256);
  const dim3 gB(128);
  const dim3 gridD (1, RTOT / 16);            // N=256  (4 waves x 64 cols)
  const dim3 gridF (DFFDIM / 256, RTOT / 16); // N=1024
  const dim3 gridO (1, RTOT / 16);            // N=64 (only wave 0 active)
  const dim3 atGrid(BNUM * HN, MSEQ / 16), atBlk(32);

  for (int i = 0; i < LLAY; ++i) {
    const size_t wOff = (size_t)i * DD;
    const size_t uOff = (size_t)i * DIM * DFFDIM;
    k_ln<<<lnGrid, lnBlk, 0, stream>>>(xbuf, ln1w + i * DIM, ln1b + i * DIM, xn);
    k_gemm<false,false,true ,false><<<gridD, gB, 0, stream>>>(xn, wtq + wOff, bq + i * DIM,  nullptr, nullptr, qb, RTOT, DIM, DIM);
    k_gemm<false,false,true ,false><<<gridD, gB, 0, stream>>>(xn, wtk + wOff, bk + i * DIM,  nullptr, nullptr, kb, RTOT, DIM, DIM);
    k_gemm<false,false,true ,true ><<<gridD, gB, 0, stream>>>(xn, wtv + wOff, bv + i * DIM,  nullptr, nullptr, vt, RTOT, DIM, DIM);
    k_attn<<<atGrid, atBlk, 0, stream>>>(qb, kb, vt, mask, att);
    k_gemm<false,true ,false,false><<<gridD, gB, 0, stream>>>(att, wto + wOff, bo + i * DIM, xbuf, xbuf, nullptr, RTOT, DIM, DIM);
    k_ln<<<lnGrid, lnBlk, 0, stream>>>(xbuf, ln2w + i * DIM, ln2b + i * DIM, xn);
    k_gemm<true ,false,true ,false><<<gridF, gB, 0, stream>>>(xn, wtu + uOff, bup + i * DFFDIM, nullptr, nullptr, hdn, RTOT, DIM, DFFDIM);
    k_gemm<false,true ,false,false><<<gridD, gB, 0, stream>>>(hdn, wtd + uOff, bdn + i * DIM, xbuf, xbuf, nullptr, RTOT, DFFDIM, DIM);
  }
  // final LN + projection D->64 straight into d_out
  k_ln<<<lnGrid, lnBlk, 0, stream>>>(xbuf, lnfw, lnfb, xn);
  k_gemm<false,false,false,false><<<gridO, gB, 0, stream>>>(xn, wtf, bfin, nullptr, out_x, nullptr, RTOT, DIM, 64);
  // score head: WMMA per-head scores + LN over heads, then in-place symmetrize
  {
    const dim3 scGrid(BNUM * (MSEQ / 16), MSEQ / 16), scBlk(32);
    k_scores<<<scGrid, scBlk, 0, stream>>>(qb, kb, snw, snb, sfw, sfb, out_s);
    size_t tot = (size_t)BNUM * MSEQ * MSEQ;
    k_sym<<<dim3((unsigned)((tot + TPB - 1) / TPB)), TPB, 0, stream>>>(out_s);
  }
}
